// LatticeLSTM_25555055411310
// MI455X (gfx1250) — matrix operand order repaired and checked by reading, not compiled, not run on hardware
//
#include <hip/hip_runtime.h>
#include <hip/hip_bf16.h>

typedef __attribute__((ext_vector_type(16))) __bf16 v16bf;
typedef __attribute__((ext_vector_type(8)))  float  v8f;
typedef __attribute__((ext_vector_type(4)))  __bf16 v4bf;   // 8-byte packed weight quad

#define T_LEN 2048
#define HID   512
#define H3    1536
#define CHUNK 4      // k-rows per async-staged chunk

// ---------------------------------------------------------------------------
// CDNA5 async-load-to-LDS helpers (ASYNCcnt-tracked DMA, ISA ch.10 / 15.18.3)
// ---------------------------------------------------------------------------
__device__ __forceinline__ void async_copy_b64(unsigned lds_off, const void* gptr) {
  asm volatile("global_load_async_to_lds_b64 %0, %1, off"
               :: "v"(lds_off), "v"((unsigned long long)gptr)
               : "memory");
}
__device__ __forceinline__ void wait_async_le8()  { asm volatile("s_wait_asynccnt 0x8" ::: "memory"); }
__device__ __forceinline__ void wait_async_le0()  { asm volatile("s_wait_asynccnt 0x0" ::: "memory"); }
__device__ __forceinline__ void wait_ds0()        { asm volatile("s_wait_dscnt 0x0"    ::: "memory"); }

// ---------------------------------------------------------------------------
// Staging kernels: fp32 -> bf16 (plus gather / transpose / pack variants)
// ---------------------------------------------------------------------------
__global__ void cvt_bf16_kernel(const float* __restrict__ in,
                                __bf16* __restrict__ out, int n) {
  int i = blockIdx.x * blockDim.x + threadIdx.x;
  if (i < n) out[i] = (__bf16)in[i];
}

// in: [K,N] row-major fp32   ->   out: [N,K] row-major bf16 (i.e. transposed)
__global__ void transpose_cvt_kernel(const float* __restrict__ in,
                                     __bf16* __restrict__ out, int K, int N) {
  int i = blockIdx.x * blockDim.x + threadIdx.x;
  if (i < K * N) {
    int n = i / K, k = i % K;
    out[i] = (__bf16)in[(size_t)k * N + n];
  }
}

// out[r*h + c] = bf16(emb[wid[r]*h + c]),   r in [0, rows)
__global__ void gather_emb_kernel(const float* __restrict__ emb,
                                  const int* __restrict__ wid,
                                  __bf16* __restrict__ out, int rows, int h) {
  int i = blockIdx.x * blockDim.x + threadIdx.x;
  if (i < rows * h) {
    int r = i / h, c = i % h;
    out[i] = (__bf16)emb[(size_t)wid[r] * h + c];
  }
}

// Pack recurrent weights into bf16 quads: out[k*HID + j] = {W[k, j], W[k, H+j],
// W[k, 2H+j], A ? A[k, j] : 0}. One 8B load then replaces 4 fp32 loads.
__global__ void pack_whh_kernel(const float* __restrict__ W,
                                const float* __restrict__ A,
                                v4bf* __restrict__ out) {
  int i = blockIdx.x * blockDim.x + threadIdx.x;
  if (i < HID * HID) {
    int k = i / HID, j = i % HID;
    v4bf w;
    w[0] = (__bf16)W[(size_t)k * H3 + j];
    w[1] = (__bf16)W[(size_t)k * H3 + HID + j];
    w[2] = (__bf16)W[(size_t)k * H3 + 2 * HID + j];
    w[3] = A ? (__bf16)A[(size_t)k * HID + j] : (__bf16)0.0f;
    out[i] = w;
  }
}

// ---------------------------------------------------------------------------
// bf16 WMMA GEMM:  C[M,N] = A[M,K] * Bt[N,K]^T + bias[N]
// One wave per 32x32 output block (2x2 register-tiled 16x16 WMMA tiles).
// ---------------------------------------------------------------------------
__global__ void gemm_bf16_wmma_kernel(const __bf16* __restrict__ A,
                                      const __bf16* __restrict__ Bt,
                                      const float* __restrict__ bias,
                                      float* __restrict__ C,
                                      int M, int N, int K) {
  int wave  = (int)((blockIdx.x * blockDim.x + threadIdx.x) >> 5);
  int lane  = threadIdx.x & 31;
  int bTn   = N >> 5;
  int nJobs = (M >> 5) * bTn;
  if (wave >= nJobs) return;           // whole-wave guard: EXEC stays all-1s
  int bm = wave / bTn, bn = wave % bTn;
  int half = lane >> 4, l16 = lane & 15;

  const __bf16* arow0 = A  + (size_t)(bm * 32 + l16) * K;
  const __bf16* arow1 = arow0 + (size_t)16 * K;
  const __bf16* bcol0 = Bt + (size_t)(bn * 32 + l16) * K;
  const __bf16* bcol1 = bcol0 + (size_t)16 * K;

  v8f acc00 = {}, acc01 = {}, acc10 = {}, acc11 = {};

  for (int kb = 0; kb < K; kb += 32) {
    v16bf a0, a1, b0, b1;
#pragma unroll
    for (int e = 0; e < 8; ++e) {
      a0[e]     = arow0[kb + half * 16 + e];
      a0[8 + e] = arow0[kb + half * 16 + 8 + e];
      a1[e]     = arow1[kb + half * 16 + e];
      a1[8 + e] = arow1[kb + half * 16 + 8 + e];
    }
#pragma unroll
    for (int e = 0; e < 16; ++e) {
      b0[e] = bcol0[kb + half * 16 + e];
      b1[e] = bcol1[kb + half * 16 + e];
    }

    if (kb + 32 < K) {                 // global_prefetch_b8 next K-panel
      __builtin_prefetch(arow0 + kb + 32, 0, 3);
      __builtin_prefetch(arow1 + kb + 32, 0, 3);
      __builtin_prefetch(bcol0 + kb + 32, 0, 3);
      __builtin_prefetch(bcol1 + kb + 32, 0, 3);
    }

    acc00 = __builtin_amdgcn_wmma_f32_16x16x32_bf16(false, a0, false, b0,
                                                    (short)0, acc00, false, false);
    acc01 = __builtin_amdgcn_wmma_f32_16x16x32_bf16(false, a0, false, b1,
                                                    (short)0, acc01, false, false);
    acc10 = __builtin_amdgcn_wmma_f32_16x16x32_bf16(false, a1, false, b0,
                                                    (short)0, acc10, false, false);
    acc11 = __builtin_amdgcn_wmma_f32_16x16x32_bf16(false, a1, false, b1,
                                                    (short)0, acc11, false, false);
  }

  int col0 = bn * 32 + l16;
  int col1 = col0 + 16;
  float bv0 = bias ? bias[col0] : 0.0f;
  float bv1 = bias ? bias[col1] : 0.0f;
#pragma unroll
  for (int r = 0; r < 8; ++r) {
    int row0 = bm * 32 + half * 8 + r;
    int row1 = row0 + 16;
    C[(size_t)row0 * N + col0] = acc00[r] + bv0;
    C[(size_t)row0 * N + col1] = acc01[r] + bv1;
    C[(size_t)row1 * N + col0] = acc10[r] + bv0;
    C[(size_t)row1 * N + col1] = acc11[r] + bv1;
  }
}

// ---------------------------------------------------------------------------
// Sequential lattice-LSTM scan. One persistent 512-thread workgroup; thread j
// owns hidden index j. The packed bf16 weight stream is double-buffered into
// LDS with GLOBAL_LOAD_ASYNC_TO_LDS_B64 (ASYNCcnt) so the FMA loop reads LDS
// while the next chunk's DMA is in flight. Each thread consumes only the LDS
// bytes it staged itself -> per-wave counter waits suffice, no extra barriers.
// ---------------------------------------------------------------------------
__global__ void __launch_bounds__(HID) lattice_scan_kernel(
    const float* __restrict__ gx,   // [T, 3H]  x@W_ih + b
    const float* __restrict__ ax,   // [T, H]   x@aW_ih + ab
    const float* __restrict__ wx,   // [2T, 3H] emb[wid]@wW_ih + wb
    const v4bf* __restrict__ P1,    // [H,H] quads {Whh_i,Whh_o,Whh_g,aWhh}
    const v4bf* __restrict__ P2,    // [H,H] quads {wWhh_f,wWhh_i,wWhh_g,0}
    float* __restrict__ hs, float* __restrict__ cs) {
  __shared__ float sh[HID];
  __shared__ float sbuf[2][2][HID];   // [slot][entry][H]; slot 0 lands this step
  __shared__ int   smask[2][2];
  __shared__ v4bf  stage[2][CHUNK][HID];   // async staging, 32 KB

  int j = threadIdx.x;
  sh[j] = 0.f;
  sbuf[0][0][j] = 0.f; sbuf[0][1][j] = 0.f;
  sbuf[1][0][j] = 0.f; sbuf[1][1][j] = 0.f;
  if (j == 0) { smask[0][0] = 0; smask[0][1] = 0; smask[1][0] = 0; smask[1][1] = 0; }
  float c = 0.f;
  unsigned sbase = (unsigned)(unsigned long long)(&stage[0][0][0]);
  __syncthreads();

  for (int t = 0; t < T_LEN; ++t) {
    // ================= phase 1: gates + alpha =================
    const float* gxt = gx + (size_t)t * H3;
    float gi = gxt[j], go = gxt[HID + j], gg = gxt[2 * HID + j];
    float a0 = ax[(size_t)t * HID + j];
    float a1 = a0;

    wait_ds0();                               // staged LDS fully consumed
#pragma unroll
    for (int kk = 0; kk < CHUNK; ++kk)        // prologue: chunk 0 in flight
      async_copy_b64(sbase + (unsigned)((kk * HID + j) * sizeof(v4bf)),
                     &P1[kk * HID + j]);

    for (int ch = 0; ch < HID / CHUNK; ++ch) {
      int buf   = ch & 1;
      int knext = (ch + 1) * CHUNK;
      if (knext < HID) {
        wait_ds0();                           // don't overwrite unread buffer
#pragma unroll
        for (int kk = 0; kk < CHUNK; ++kk)
          async_copy_b64(sbase + (unsigned)((((buf ^ 1) * CHUNK + kk) * HID + j) * sizeof(v4bf)),
                         &P1[(knext + kk) * HID + j]);
        wait_async_le8();                     // <=8 outstanding: both chunks may be in flight
      } else {
        wait_async_le0();
      }
#pragma unroll
      for (int kk = 0; kk < CHUNK; ++kk) {
        int k = ch * CHUNK + kk;
        float hk = sh[k];
        v4bf w = stage[buf][kk][j];
        gi += hk * (float)w[0];
        go += hk * (float)w[1];
        gg += hk * (float)w[2];
        float av = (float)w[3];
        a0 += sbuf[0][0][k] * av;
        a1 += sbuf[0][1][k] * av;
      }
    }

    float ig = 1.f / (1.f + expf(-gi));
    float og = 1.f / (1.f + expf(-go));
    float gt = tanhf(gg);

    int m0 = smask[0][0], m1 = smask[0][1];
    float al0 = 1.f / (1.f + expf(-a0));
    float al1 = 1.f / (1.f + expf(-a1));
    float e_i  = expf(ig);
    float e_a0 = m0 ? expf(al0) : 0.f;
    float e_a1 = m1 ? expf(al1) : 0.f;
    float denom   = e_i + e_a0 + e_a1;
    float c_merge = (e_i * gt + e_a0 * sbuf[0][0][j] + e_a1 * sbuf[0][1][j]) / denom;
    float c_plain = (1.f - ig) * c + ig * gt;
    float c1 = (m0 || m1) ? c_merge : c_plain;
    float h1 = og * tanhf(c1);

    hs[(size_t)t * HID + j] = h1;
    cs[(size_t)t * HID + j] = c1;
    c = c1;

    __syncthreads();
    sh[j] = h1;
    __syncthreads();

    // ================= phase 2: WordLSTM hidden matvec =================
    float hf = 0.f, hi = 0.f, hg = 0.f;

    wait_ds0();
#pragma unroll
    for (int kk = 0; kk < CHUNK; ++kk)
      async_copy_b64(sbase + (unsigned)((kk * HID + j) * sizeof(v4bf)),
                     &P2[kk * HID + j]);

    for (int ch = 0; ch < HID / CHUNK; ++ch) {
      int buf   = ch & 1;
      int knext = (ch + 1) * CHUNK;
      if (knext < HID) {
        wait_ds0();
#pragma unroll
        for (int kk = 0; kk < CHUNK; ++kk)
          async_copy_b64(sbase + (unsigned)((((buf ^ 1) * CHUNK + kk) * HID + j) * sizeof(v4bf)),
                         &P2[(knext + kk) * HID + j]);
        wait_async_le8();
      } else {
        wait_async_le0();
      }
#pragma unroll
      for (int kk = 0; kk < CHUNK; ++kk) {
        int k = ch * CHUNK + kk;
        float hk = sh[k];
        v4bf w = stage[buf][kk][j];
        hf += hk * (float)w[0];
        hi += hk * (float)w[1];
        hg += hk * (float)w[2];
      }
    }

    const float* wx0 = wx + (size_t)(2 * t) * H3;
    const float* wx1 = wx0 + H3;
    float f0 = wx0[j] + hf, i0 = wx0[HID + j] + hi, g0 = wx0[2 * HID + j] + hg;
    float f1 = wx1[j] + hf, i1 = wx1[HID + j] + hi, g1 = wx1[2 * HID + j] + hg;
    float ct0 = (1.f / (1.f + expf(-f0))) * c1 + (1.f / (1.f + expf(-i0))) * tanhf(g0);
    float ct1 = (1.f / (1.f + expf(-f1))) * c1 + (1.f / (1.f + expf(-i1))) * tanhf(g1);

    // ---- ring-buffer shift ----
    float old11 = sbuf[1][1][j];
    int oldm11  = smask[1][1];
    __syncthreads();
    sbuf[0][0][j] = ct0;      // len-2 match lands at t+1
    sbuf[0][1][j] = old11;    // len-3 match deposited last step lands next step
    sbuf[1][0][j] = 0.f;
    sbuf[1][1][j] = ct1;      // len-3 match lands at t+2
    if (j == 0) {
      smask[0][0] = (t < T_LEN - 1) ? 1 : 0;
      smask[0][1] = oldm11;
      smask[1][0] = 0;
      smask[1][1] = (t < T_LEN - 2) ? 1 : 0;
    }
    __syncthreads();
  }
}

// ---------------------------------------------------------------------------
// Launch
// ---------------------------------------------------------------------------
extern "C" void kernel_launch(void* const* d_in, const int* in_sizes, int n_in,
                              void* d_out, int out_size, void* d_ws, size_t ws_size,
                              hipStream_t stream) {
  (void)in_sizes; (void)n_in; (void)out_size; (void)ws_size;

  const float* x     = (const float*)d_in[0];
  const int*   wid   = (const int*)d_in[1];
  const float* W_ih  = (const float*)d_in[2];
  const float* W_hh  = (const float*)d_in[3];
  const float* b     = (const float*)d_in[4];
  const float* aW_ih = (const float*)d_in[5];
  const float* aW_hh = (const float*)d_in[6];
  const float* ab    = (const float*)d_in[7];
  const float* wW_ih = (const float*)d_in[8];
  const float* wW_hh = (const float*)d_in[9];
  const float* wb    = (const float*)d_in[10];
  const float* emb   = (const float*)d_in[11];

  // workspace layout
  float* gx = (float*)d_ws;                                 // [T, 3H]
  float* ax = gx + (size_t)T_LEN * H3;                      // [T, H]
  float* wx = ax + (size_t)T_LEN * HID;                     // [2T, 3H]
  __bf16* xbf   = (__bf16*)(wx + (size_t)T_LEN * 2 * H3);   // [T, H]
  __bf16* webf  = xbf   + (size_t)T_LEN * HID;              // [2T, H]
  __bf16* WihT  = webf  + (size_t)2 * T_LEN * HID;          // [3H, H]
  __bf16* aWihT = WihT  + (size_t)H3 * HID;                 // [H, H]
  __bf16* wWihT = aWihT + (size_t)HID * HID;                // [3H, H]
  v4bf*   pack1 = (v4bf*)(wWihT + (size_t)H3 * HID);        // [H*H] quads
  v4bf*   pack2 = pack1 + (size_t)HID * HID;                // [H*H] quads

  // stage bf16 operands
  int nx = T_LEN * HID;
  cvt_bf16_kernel<<<(nx + 255) / 256, 256, 0, stream>>>(x, xbf, nx);
  int ne = 2 * T_LEN * HID;
  gather_emb_kernel<<<(ne + 255) / 256, 256, 0, stream>>>(emb, wid, webf, 2 * T_LEN, HID);
  int nw = HID * H3;
  transpose_cvt_kernel<<<(nw + 255) / 256, 256, 0, stream>>>(W_ih, WihT, HID, H3);
  int na = HID * HID;
  transpose_cvt_kernel<<<(na + 255) / 256, 256, 0, stream>>>(aW_ih, aWihT, HID, HID);
  transpose_cvt_kernel<<<(nw + 255) / 256, 256, 0, stream>>>(wW_ih, wWihT, HID, H3);
  pack_whh_kernel<<<(na + 255) / 256, 256, 0, stream>>>(W_hh, aW_hh, pack1);
  pack_whh_kernel<<<(na + 255) / 256, 256, 0, stream>>>(wW_hh, nullptr, pack2);

  // parallel input-side GEMMs (WMMA bf16): 1 wave = one 32x32 block,
  // 8 waves per 256-thread block
  {
    int jobs = (T_LEN / 32) * (H3 / 32);      // 3072
    gemm_bf16_wmma_kernel<<<jobs / 8, 256, 0, stream>>>(xbf, WihT, b, gx, T_LEN, H3, HID);
  }
  {
    int jobs = (T_LEN / 32) * (HID / 32);     // 1024
    gemm_bf16_wmma_kernel<<<jobs / 8, 256, 0, stream>>>(xbf, aWihT, ab, ax, T_LEN, HID, HID);
  }
  {
    int jobs = (2 * T_LEN / 32) * (H3 / 32);  // 6144
    gemm_bf16_wmma_kernel<<<jobs / 8, 256, 0, stream>>>(webf, wWihT, wb, wx, 2 * T_LEN, H3, HID);
  }

  // serial recurrence
  float* hs = (float*)d_out;
  float* cs = hs + (size_t)T_LEN * HID;
  lattice_scan_kernel<<<1, HID, 0, stream>>>(gx, ax, wx, pack1, pack2, hs, cs);
}
